// DeformBlock_77163382440850
// MI455X (gfx1250) — matrix operand order repaired and checked by reading, not compiled, not run on hardware
//
#include <hip/hip_runtime.h>
#include <math.h>

typedef __attribute__((ext_vector_type(16))) _Float16     v16h;
typedef __attribute__((ext_vector_type(8)))  float        v8f;
typedef __attribute__((ext_vector_type(4)))  unsigned int v4u;
typedef __attribute__((ext_vector_type(8)))  int          v8i;
typedef __attribute__((ext_vector_type(4)))  int          v4i;

#ifndef __has_builtin
#define __has_builtin(x) 0
#endif

// Problem constants (B,C,H,W fixed by the harness reference)
#define BN   8
#define CC   64
#define HH   128
#define WW   128
#define HWSZ (HH * WW)        // 16384
#define K2   9
#define KTOT (CC * K2)        // 576
#define N9   (BN * HWSZ * K2) // 1179648 floats per offset field
#define DWH_BYTES (CC * KTOT * 2)   // 73728 bytes of f16 weights
#define DWH_DWORDS (DWH_BYTES / 4)  // 18432

// ---------------------------------------------------------------------------
// Kernel 0: repack dw[O][C][3][3] (f32) -> dwh[n][kt] (f16), kt = t*64 + c.
// Tap-major K ordering => each 32-wide WMMA K-step covers a single tap t.
// ---------------------------------------------------------------------------
__global__ __launch_bounds__(256) void reorder_dw_kernel(
    const float* __restrict__ dw, _Float16* __restrict__ dwh)
{
    int i = blockIdx.x * 256 + threadIdx.x;          // 0 .. 36863
    if (i < CC * KTOT) {
        int n = i / KTOT;
        int kt = i - n * KTOT;
        int t = kt >> 6;                              // tap
        int c = kt & 63;                              // channel
        dwh[i] = (_Float16)dw[((size_t)n * CC + c) * K2 + t];
    }
}

// ---------------------------------------------------------------------------
// Kernel 1: offset/mask 3x3 conv (SAME), LDS-tiled 16x16 pixel blocks.
// Outputs pixel-major dy/dx/mask arrays [B*H*W][9] for contiguous reads.
// ---------------------------------------------------------------------------
__global__ __launch_bounds__(256) void offset_mask_kernel(
    const float* __restrict__ x,
    const float* __restrict__ ow, const float* __restrict__ ob,
    const float* __restrict__ mw, const float* __restrict__ mb,
    float* __restrict__ dyw, float* __restrict__ dxw, float* __restrict__ mkw)
{
    __shared__ float patch[18 * 18];

    const int bx = blockIdx.x & 7;
    const int by = (blockIdx.x >> 3) & 7;
    const int b  = blockIdx.x >> 6;
    const int tx = threadIdx.x & 15;
    const int ty = threadIdx.x >> 4;
    const int gx0 = bx * 16 - 1;
    const int gy0 = by * 16 - 1;

    float acc[27];
#pragma unroll
    for (int oc = 0; oc < 18; ++oc) acc[oc] = ob[oc];
#pragma unroll
    for (int t = 0; t < 9; ++t)     acc[18 + t] = mb[t];

    const float* xb = x + (size_t)b * CC * HWSZ;

#pragma unroll 1
    for (int c = 0; c < CC; ++c) {
        // cooperative load of 18x18 zero-padded patch for channel c
        for (int i = threadIdx.x; i < 18 * 18; i += 256) {
            int pyy = i / 18, pxx = i - pyy * 18;
            int gy = gy0 + pyy, gx = gx0 + pxx;
            bool ok = (gy >= 0) & (gy < HH) & (gx >= 0) & (gx < WW);
            patch[i] = ok ? xb[(size_t)c * HWSZ + gy * WW + gx] : 0.0f;
        }
        __syncthreads();
#pragma unroll
        for (int tap = 0; tap < 9; ++tap) {
            const float v = patch[(ty + tap / 3) * 18 + tx + (tap % 3)];
#pragma unroll
            for (int oc = 0; oc < 18; ++oc)
                acc[oc] += v * ow[((size_t)oc * CC + c) * 9 + tap];
#pragma unroll
            for (int t = 0; t < 9; ++t)
                acc[18 + t] += v * mw[((size_t)t * CC + c) * 9 + tap];
        }
        __syncthreads();
    }

    const size_t P = (size_t)b * HWSZ + (by * 16 + ty) * WW + bx * 16 + tx;
#pragma unroll
    for (int t = 0; t < 9; ++t) {
        dyw[P * 9 + t] = acc[2 * t + 0];              // even channel = dy
        dxw[P * 9 + t] = acc[2 * t + 1];              // odd  channel = dx
        mkw[P * 9 + t] = 1.0f / (1.0f + __expf(-acc[18 + t]));
    }
}

// ---------------------------------------------------------------------------
// Kernel 2: fused bilinear deformable sampling + WMMA GEMM.
// One wave: 16 pixels (M) x 64 out-channels (4 N-tiles), K = 576 (tap-major).
// B operand (72KB f16) staged LDS-resident per workgroup via the Tensor Data
// Mover (tensor_load_to_lds + s_wait_tensorcnt), shared by all 8 waves.
// ---------------------------------------------------------------------------
__global__ __launch_bounds__(256) void deform_wmma_kernel(
    const float*    __restrict__ x,
    const _Float16* __restrict__ dwh,
    const float*    __restrict__ db,
    const float*    __restrict__ dyw,
    const float*    __restrict__ dxw,
    const float*    __restrict__ mkw,
    float*          __restrict__ out)
{
    extern __shared__ _Float16 ldsw[];   // 36864 halves = 72KB (dynamic LDS)

    // ---- stage dwh -> LDS once per workgroup -------------------------------
#if __has_builtin(__builtin_amdgcn_tensor_load_to_lds)
    if (threadIdx.x < 32) {              // TDM ignores EXEC: issue on 1 wave
        const unsigned long long ga = (unsigned long long)(uintptr_t)dwh;
        const unsigned int lds_base =
            (unsigned int)__builtin_amdgcn_groupstaticsize();
        // D# group0: count=1 | lds_addr | global_addr[56:0] | type=2 (ISA 8.3)
        v4u g0;
        g0.x = 1u;                                          // count=1, user D#
        g0.y = lds_base;                                    // lds_addr (bytes)
        g0.z = (unsigned int)ga;                            // global_addr lo
        g0.w = (unsigned int)((ga >> 32) & 0x01FFFFFFu) | (2u << 30); // type=2
        // D# group1: data_size=4B, 1-D tile of 18432 dwords (ISA 8.4)
        v8i g1;
        g1[0] = (int)(2u << 16);                 // data_size = 2 (4 bytes)
        g1[1] = (int)((DWH_DWORDS & 0xFFFFu) << 16);  // tensor_dim0[15:0]
        g1[2] = (int)(1u << 16);                 // tensor_dim0 hi=0, dim1=1
        g1[3] = (int)((DWH_DWORDS & 0xFFFFu) << 16);  // tile_dim0 = 18432
        g1[4] = 1;                               // tile_dim1 = 1
        g1[5] = DWH_DWORDS;                      // tensor_dim0_stride
        g1[6] = 0;
        g1[7] = 0;
        const v4i gz4 = {0, 0, 0, 0};            // groups 2/3 unused (<=2-D)
        const v8i gz8 = {0, 0, 0, 0, 0, 0, 0, 0};
        __builtin_amdgcn_tensor_load_to_lds(g0, g1, gz4, gz4, gz8, 0);
        __builtin_amdgcn_s_wait_tensorcnt(0);
    }
#else
    for (int i = threadIdx.x; i < CC * KTOT; i += 256) ldsw[i] = dwh[i];
#endif
    __syncthreads();

    const int lane = threadIdx.x & 31;
    const int wave = threadIdx.x >> 5;
    const int tile = blockIdx.x * 8 + wave;           // 8192 tiles total
    const int m    = lane & 15;                       // pixel-in-tile (M)
    const int P0   = tile * 16;
    const int b    = P0 >> 14;                        // / HWSZ
    const int pixBase = P0 & (HWSZ - 1);              // uniform in wave
    const int P    = P0 + m;
    const int pix  = pixBase + m;
    const int h    = pix >> 7;
    const int w    = pix & 127;
    const float* xb = x + (size_t)b * CC * HWSZ;
    const int laneAddA = (lane >= 16) ? 8 : 0;        // ISA 16-bit A layout
    const int laneAddB = (lane >= 16) ? 16 : 0;       // B half-wave K offset

    v8f acc[4] = {};

#pragma unroll 1
    for (int t = 0; t < 9; ++t) {
        // ---- per-tap bilinear setup (mask folded into corner weights) ----
        const size_t o9 = (size_t)P * 9 + t;
        const float py = (float)(h + (t / 3) - 1) + dyw[o9];
        const float px = (float)(w + (t % 3) - 1) + dxw[o9];
        const float mk = mkw[o9];
        const float y0f = floorf(py), x0f = floorf(px);
        const int   y0 = (int)y0f,  x0 = (int)x0f;
        const int   y1 = y0 + 1,    x1 = x0 + 1;
        const float ly = py - y0f,  lx = px - x0f;
        const float hy = 1.0f - ly, hx = 1.0f - lx;
        const bool vy0 = (y0 >= 0) & (y0 < HH), vy1 = (y1 >= 0) & (y1 < HH);
        const bool vx0 = (x0 >= 0) & (x0 < WW), vx1 = (x1 >= 0) & (x1 < WW);
        const int cy0 = min(max(y0, 0), HH - 1), cy1 = min(max(y1, 0), HH - 1);
        const int cx0 = min(max(x0, 0), WW - 1), cx1 = min(max(x1, 0), WW - 1);
        const int i00 = cy0 * WW + cx0, i01 = cy0 * WW + cx1;
        const int i10 = cy1 * WW + cx0, i11 = cy1 * WW + cx1;
        const float w00 = (vy0 & vx0) ? hy * hx * mk : 0.0f;
        const float w01 = (vy0 & vx1) ? hy * lx * mk : 0.0f;
        const float w10 = (vy1 & vx0) ? ly * hx * mk : 0.0f;
        const float w11 = (vy1 & vx1) ? ly * lx * mk : 0.0f;

        // ---- two 32-wide K-steps cover channels 0..31 / 32..63 of tap t ----
#pragma unroll
        for (int sub = 0; sub < 2; ++sub) {
            // Build A fragment in contiguous K order (matches B):
            //   K(e,lane) = e + 16*(lane>=16)
            v16h a;
#pragma unroll
            for (int e = 0; e < 16; ++e) {
                const int c = sub * 32 + e + laneAddB;
                const float* pl = xb + (size_t)c * HWSZ;
                const float v = w00 * pl[i00] + w01 * pl[i01]
                              + w10 * pl[i10] + w11 * pl[i11];
                a[e] = (_Float16)v;
            }
            const int kb = (t * 2 + sub) * 32 + laneAddB;
#pragma unroll
            for (int j = 0; j < 4; ++j) {
                // B fragment: 16 contiguous f16 (32B aligned) from LDS
                const v16h bf =
                    *(const v16h*)(ldsw + (size_t)(j * 16 + m) * KTOT + kb);
                acc[j] = __builtin_amdgcn_wmma_f32_16x16x32_f16(
                    false, a, false, bf, (short)0, acc[j], false, false);
            }
        }
    }

    // ---- epilogue: D layout (lane<16: M=r, N=lane; lane>=16: M=8+r) ----
    const int laneHi = lane >> 4;
#pragma unroll
    for (int j = 0; j < 4; ++j) {
        const int o = j * 16 + m;
        const float bias = db[o];
        float* op = out + ((size_t)(b * CC + o) << 14) + pixBase;
#pragma unroll
        for (int r = 0; r < 8; ++r)
            op[r + 8 * laneHi] = acc[j][r] + bias;
    }
}

// ---------------------------------------------------------------------------
// Host launcher
// ---------------------------------------------------------------------------
extern "C" void kernel_launch(void* const* d_in, const int* in_sizes, int n_in,
                              void* d_out, int out_size, void* d_ws, size_t ws_size,
                              hipStream_t stream)
{
    const float* x  = (const float*)d_in[0];
    const float* ow = (const float*)d_in[1];
    const float* ob = (const float*)d_in[2];
    const float* mw = (const float*)d_in[3];
    const float* mb = (const float*)d_in[4];
    const float* dw = (const float*)d_in[5];
    const float* db = (const float*)d_in[6];
    float* out = (float*)d_out;

    // Workspace partition: dy, dx, mask (f32, pixel-major [P][9]) + dwh (f16)
    float*    dyw = (float*)d_ws;
    float*    dxw = dyw + N9;
    float*    mkw = dxw + N9;
    _Float16* dwh = (_Float16*)(mkw + N9);   // 36864 halves, 32B-aligned offset

    // Allow 72KB of dynamic LDS for the WMMA kernel (idempotent each call).
    (void)hipFuncSetAttribute((const void*)deform_wmma_kernel,
                              hipFuncAttributeMaxDynamicSharedMemorySize,
                              DWH_BYTES);

    // 0) repack deformable conv weights to f16 tap-major K order
    reorder_dw_kernel<<<(CC * KTOT + 255) / 256, 256, 0, stream>>>(dw, dwh);

    // 1) offset / mask 3x3 conv + sigmoid  (512 blocks = B * 8 * 8 tiles)
    offset_mask_kernel<<<BN * (HH / 16) * (WW / 16), 256, 0, stream>>>(
        x, ow, ob, mw, mb, dyw, dxw, mkw);

    // 2) fused sampling + WMMA GEMM: 8192 wave-tiles / 8 waves per block
    deform_wmma_kernel<<<(BN * HWSZ / 16) / 8, 256, DWH_BYTES, stream>>>(
        x, dwh, db, dyw, dxw, mkw, out);
}
